// GraphAttention_25838523252953
// MI455X (gfx1250) — matrix operand order repaired and checked by reading, not compiled, not run on hardware
//
#include <hip/hip_runtime.h>
#include <hip/hip_bf16.h>

typedef __attribute__((ext_vector_type(16))) _Float16 v16h;
typedef __attribute__((ext_vector_type(8)))  _Float16 v8h;
typedef __attribute__((ext_vector_type(8)))  float    v8f;

#define IN_DIM 128
#define N_HEADS 4
#define HEAD_DIM 32
#define INV_SQRT_HD 0.17677669529663687f  // 1/sqrt(32)
#define W_ELEMS (IN_DIM * IN_DIM)         // 16384

// ---------- order-preserving float<->uint key (for atomic segment max) ------
__device__ __forceinline__ unsigned fkey(float f) {
    unsigned u = __float_as_uint(f);
    return (u & 0x80000000u) ? ~u : (u | 0x80000000u);
}
__device__ __forceinline__ float funkey(unsigned kk) {
    unsigned u = (kk & 0x80000000u) ? (kk & 0x7FFFFFFFu) : ~kk;
    return __uint_as_float(u);
}

// ---------------------------------------------------------------------------
// Kernel 0b: one-shot f32 -> f16 conversion of the 4 weight matrices.
// Removes all v_cvt from the WMMA inner loops and halves B-fragment bytes.
// ---------------------------------------------------------------------------
__global__ void __launch_bounds__(256)
convert_weights(const float* __restrict__ Wq, const float* __restrict__ Wk,
                const float* __restrict__ Wv, const float* __restrict__ Wo,
                _Float16* __restrict__ Wh)
{
    const int idx = blockIdx.x * 256 + threadIdx.x;
    if (idx >= 4 * W_ELEMS) return;
    const int mat = idx >> 14;
    const int off = idx & (W_ELEMS - 1);
    const float* W = (mat == 0) ? Wq : (mat == 1) ? Wk : (mat == 2) ? Wv : Wo;
    Wh[idx] = (_Float16)W[off];
}

// ---------------------------------------------------------------------------
// Fragment loaders (contiguous k order, the same for A and B).
// A (16-bit, 16x32): lane L, elem e -> K = e + 16*(L>=16), M=L%16
//   => two contiguous 16B LDS loads + shuffle.
// B (16-bit, 32x16): lane L, elem e -> K = e + 16*(L>=16), N=L%16
//   => one contiguous 32B load (B[k][n] = W[n][k], row-major W).
// ---------------------------------------------------------------------------
__device__ __forceinline__ v16h load_a_frag(const _Float16* base /*row+k0+hi*16*/) {
    v8h lo = *(const v8h*)(base);
    v8h hi = *(const v8h*)(base + 8);
    return __builtin_shufflevector(lo, hi, 0, 1, 2, 3, 4, 5, 6, 7,
                                           8, 9, 10, 11, 12, 13, 14, 15);
}

// ---------------------------------------------------------------------------
// Kernel 1: fused Q/K/V projection with WMMA.
// Block = 256 threads (8 waves), one 16-node M-tile per block.
// 24 (matrix, n-tile) combos -> 3 per wave, K=128 -> 4 wmma per combo.
// ---------------------------------------------------------------------------
__global__ void __launch_bounds__(256)
qkv_proj_wmma(const float* __restrict__ x,
              const _Float16* __restrict__ Wh,   // [3][128][128] f16 (q,k,v)
              const float* __restrict__ bq, const float* __restrict__ bk,
              const float* __restrict__ bv,
              float* __restrict__ q, float* __restrict__ k, float* __restrict__ v,
              int n_nodes)
{
    __shared__ alignas(32) _Float16 xh[16 * IN_DIM];
    const int tid  = threadIdx.x;
    const int wave = tid >> 5;
    const int lane = tid & 31;
    const int m0   = blockIdx.x * 16;

    // stage x tile (f32 -> f16), coalesced, once per block
    for (int i = tid; i < 16 * IN_DIM; i += 256) {
        int gm = m0 + (i >> 7);
        xh[i] = (gm < n_nodes) ? (_Float16)x[(size_t)gm * IN_DIM + (i & 127)] : (_Float16)0.0f;
    }
    __syncthreads();

    const int mrow = lane & 15;
    const int hi   = lane >> 4;     // 0 for lanes 0-15, 1 for lanes 16-31

    #pragma unroll
    for (int t = 0; t < 3; ++t) {
        const int combo = wave * 3 + t;          // 0..23, unique
        const int mat   = combo >> 3;            // 0=q 1=k 2=v
        const int ntile = combo & 7;             // 0..7
        const int n     = ntile * 16 + mrow;

        const _Float16* W = Wh + (size_t)mat * W_ELEMS;
        const float*    B = (mat == 0) ? bq : (mat == 1) ? bk : bv;
        float*          O = (mat == 0) ? q  : (mat == 1) ? k  : v;

        v8f acc = {};
        #pragma unroll
        for (int k0 = 0; k0 < IN_DIM; k0 += 32) {
            v16h a = load_a_frag(&xh[mrow * IN_DIM + k0 + hi * 16]);
            v16h b = *(const v16h*)&W[(size_t)n * IN_DIM + k0 + hi * 16];
            acc = __builtin_amdgcn_wmma_f32_16x16x32_f16(false, a, false, b,
                                                         (short)0, acc, false, false);
        }
        const float bias = B[n];
        #pragma unroll
        for (int r = 0; r < 8; ++r) {
            int m = m0 + r + hi * 8;
            if (m < n_nodes) O[(size_t)m * IN_DIM + n] = acc[r] + bias;
        }
    }
}

// ---------------------------------------------------------------------------
// Kernel 2: per-edge scores + atomic segment max. One wave32 per edge.
// Lane i holds 4 contiguous dims; head = lane/8; shfl_xor tree within 8 lanes.
// ---------------------------------------------------------------------------
__global__ void __launch_bounds__(256)
edge_scores_max(const long long* __restrict__ ei,
                const float* __restrict__ q, const float* __restrict__ k,
                float* __restrict__ scores, unsigned* __restrict__ segmax, int E)
{
    const int gw   = (int)((blockIdx.x * blockDim.x + threadIdx.x) >> 5);
    const int lane = threadIdx.x & 31;
    if (gw >= E) return;
    const long long row = ei[gw];
    const long long col = ei[(size_t)E + gw];

    const float4* q4 = (const float4*)q;
    const float4* k4 = (const float4*)k;
    float4 a = q4[(size_t)row * 32 + lane];
    float4 b = k4[(size_t)col * 32 + lane];
    float p = a.x * b.x + a.y * b.y + a.z * b.z + a.w * b.w;
    p += __shfl_xor(p, 1);
    p += __shfl_xor(p, 2);
    p += __shfl_xor(p, 4);            // all 8 lanes of a head now hold head-dot
    const float score = p * INV_SQRT_HD;

    if ((lane & 7) == 0) {
        const int h = lane >> 3;
        scores[(size_t)gw * N_HEADS + h] = score;
        atomicMax(&segmax[(size_t)row * N_HEADS + h], fkey(score));
    }
}

// ---------------------------------------------------------------------------
// Kernel 3: e = exp(score - segmax[row]); atomic segment sum.
// One thread per (edge, head). Overwrites scores in place (rewritten by K2
// each call, so graph replay stays deterministic).
// ---------------------------------------------------------------------------
__global__ void __launch_bounds__(256)
edge_exp_sum(const long long* __restrict__ ei, float* __restrict__ scores,
             const unsigned* __restrict__ segmax, float* __restrict__ segsum, int E)
{
    const int t = blockIdx.x * blockDim.x + threadIdx.x;
    if (t >= E * N_HEADS) return;
    const int e = t >> 2, h = t & 3;
    const long long row = ei[e];
    const float m  = funkey(segmax[(size_t)row * N_HEADS + h]);
    const float ex = __expf(scores[t] - m);
    scores[t] = ex;
    atomicAdd(&segsum[(size_t)row * N_HEADS + h], ex);
}

// ---------------------------------------------------------------------------
// Kernel 4: normalize, weighted scatter-add of v[col] into attended[row],
// and avg_attn output. One wave32 per edge.
// ---------------------------------------------------------------------------
__global__ void __launch_bounds__(256)
edge_norm_scatter(const long long* __restrict__ ei, const float* __restrict__ scores,
                  const float* __restrict__ segsum, const float* __restrict__ v,
                  float* __restrict__ attended, float* __restrict__ avg_out, int E)
{
    const int gw   = (int)((blockIdx.x * blockDim.x + threadIdx.x) >> 5);
    const int lane = threadIdx.x & 31;
    if (gw >= E) return;
    const long long row = ei[gw];
    const long long col = ei[(size_t)E + gw];

    const int h = lane >> 3;
    const float ex  = scores[(size_t)gw * N_HEADS + h];
    const float sum = segsum[(size_t)row * N_HEADS + h];
    const float a   = ex / (sum + 1e-16f);

    const float4* v4 = (const float4*)v;
    float4 vv = v4[(size_t)col * 32 + lane];
    float* dst = attended + (size_t)row * IN_DIM + lane * 4;
    atomicAdd(dst + 0, a * vv.x);
    atomicAdd(dst + 1, a * vv.y);
    atomicAdd(dst + 2, a * vv.z);
    atomicAdd(dst + 3, a * vv.w);

    // mean over heads: lanes {0,8,16,24} hold the 4 distinct attn values
    float s = __shfl(a, 0) + __shfl(a, 8) + __shfl(a, 16) + __shfl(a, 24);
    if (lane == 0) avg_out[gw] = s * 0.25f;
}

// ---------------------------------------------------------------------------
// Kernel 5: out = attended @ Wo.T + bo + x, WMMA. 8 waves -> 8 n-tiles.
// ---------------------------------------------------------------------------
__global__ void __launch_bounds__(256)
out_proj_wmma(const float* __restrict__ att, const float* __restrict__ x,
              const _Float16* __restrict__ Who, const float* __restrict__ bo,
              float* __restrict__ out, int n_nodes)
{
    __shared__ alignas(32) _Float16 ah[16 * IN_DIM];
    const int tid  = threadIdx.x;
    const int wave = tid >> 5;
    const int lane = tid & 31;
    const int m0   = blockIdx.x * 16;

    for (int i = tid; i < 16 * IN_DIM; i += 256) {
        int gm = m0 + (i >> 7);
        ah[i] = (gm < n_nodes) ? (_Float16)att[(size_t)gm * IN_DIM + (i & 127)] : (_Float16)0.0f;
    }
    __syncthreads();

    const int mrow = lane & 15;
    const int hi   = lane >> 4;
    const int n    = wave * 16 + mrow;

    v8f acc = {};
    #pragma unroll
    for (int k0 = 0; k0 < IN_DIM; k0 += 32) {
        v16h a = load_a_frag(&ah[mrow * IN_DIM + k0 + hi * 16]);
        v16h b = *(const v16h*)&Who[(size_t)n * IN_DIM + k0 + hi * 16];
        acc = __builtin_amdgcn_wmma_f32_16x16x32_f16(false, a, false, b,
                                                     (short)0, acc, false, false);
    }
    const float bias = bo[n];
    #pragma unroll
    for (int r = 0; r < 8; ++r) {
        int m = m0 + r + hi * 8;
        if (m < n_nodes)
            out[(size_t)m * IN_DIM + n] = acc[r] + bias + x[(size_t)m * IN_DIM + n];
    }
}

// ---------------------------------------------------------------------------
extern "C" void kernel_launch(void* const* d_in, const int* in_sizes, int n_in,
                              void* d_out, int out_size, void* d_ws, size_t ws_size,
                              hipStream_t stream) {
    const float*     x  = (const float*)d_in[0];
    const long long* ei = (const long long*)d_in[1];   // [2, E] int64
    const float* Wq = (const float*)d_in[2];
    const float* bq = (const float*)d_in[3];
    const float* Wk = (const float*)d_in[4];
    const float* bk = (const float*)d_in[5];
    const float* Wv = (const float*)d_in[6];
    const float* bv = (const float*)d_in[7];
    const float* Wo = (const float*)d_in[8];
    const float* bo = (const float*)d_in[9];

    const int N = in_sizes[0] / IN_DIM;   // 100000
    const int E = in_sizes[1] / 2;        // 800000

    // workspace layout (floats):
    // [attended N*128][segmax N*4 (u32)][segsum N*4][q N*128][k N*128][v N*128]
    // [scores E*4][Wh 4*16384 f16]
    float*    ws       = (float*)d_ws;
    size_t    nAtt     = (size_t)N * IN_DIM;
    float*    attended = ws;
    unsigned* segmax   = (unsigned*)(ws + nAtt);
    float*    segsum   = ws + nAtt + (size_t)N * N_HEADS;
    float*    q        = segsum + (size_t)N * N_HEADS;
    float*    k        = q + nAtt;
    float*    v        = k + nAtt;
    float*    scores   = v + nAtt;
    _Float16* Wh       = (_Float16*)(scores + (size_t)E * N_HEADS);  // 32B-aligned
    _Float16* Who      = Wh + (size_t)3 * W_ELEMS;

    float* out_nodes = (float*)d_out;            // [N,128]
    float* out_avg   = (float*)d_out + nAtt;     // [E]

    // zero attended / segmax keys / segsum (all-zero bit patterns are correct inits)
    hipMemsetAsync(d_ws, 0, (nAtt + (size_t)N * N_HEADS * 2) * sizeof(float), stream);

    convert_weights<<<(4 * W_ELEMS + 255) / 256, 256, 0, stream>>>(Wq, Wk, Wv, Wo, Wh);

    const int tiles = (N + 15) / 16;
    qkv_proj_wmma<<<tiles, 256, 0, stream>>>(x, Wh, bq, bk, bv, q, k, v, N);

    const int edgeBlocks = (E + 7) / 8;          // 8 waves (edges) per 256-thread block
    edge_scores_max<<<edgeBlocks, 256, 0, stream>>>(ei, q, k, scores, segmax, E);

    const int ehBlocks = (E * N_HEADS + 255) / 256;
    edge_exp_sum<<<ehBlocks, 256, 0, stream>>>(ei, scores, segmax, segsum, E);

    edge_norm_scatter<<<edgeBlocks, 256, 0, stream>>>(ei, scores, segsum, v,
                                                      attended, out_avg, E);

    out_proj_wmma<<<tiles, 256, 0, stream>>>(attended, x, Who, bo, out_nodes, N);
}
